// AERO_41721312313595
// MI455X (gfx1250) — matrix operand order repaired and checked by reading, not compiled, not run on hardware
//
#include <hip/hip_runtime.h>
#include <math.h>

typedef __attribute__((ext_vector_type(16))) _Float16 v16h;
typedef __attribute__((ext_vector_type(8)))  float    v8f;
typedef __attribute__((ext_vector_type(4)))  float    v4f;

#define HH 8
#define CC 16

__device__ __forceinline__ float eluf(float x) {
    return x > 0.0f ? x : (expf(x) - 1.0f);
}

// ---------------------------------------------------------------------------
// Pack a row-major f32 weight [Kdim x Ncols] into WMMA-B f16 fragment order:
// fragment (kt,nt) holds B tile rows K=kt*32..+31, cols N=nt*16..+15.
// Order used here (contiguous K, not the ISA table order; A matches): lane = half*16 + (N%16),
// element e of v16h holds K = kt*32 + half*16 + e.
// Stored as Bp[((kt*ntiles + nt)*32 + lane)*16 + e] so each lane reads one
// contiguous 32-byte fragment.
// ---------------------------------------------------------------------------
__global__ void pack_b_kernel(const float* __restrict__ W, _Float16* __restrict__ Bp,
                              int Kdim, int Ncols, int ntiles) {
    int idx = blockIdx.x * blockDim.x + threadIdx.x;
    int total = (Kdim >> 5) * ntiles * 32;
    if (idx >= total) return;
    int lane = idx & 31;
    int t    = idx >> 5;
    int nt   = t % ntiles;
    int kt   = t / ntiles;
    int hl   = lane >> 4;
    int n    = nt * 16 + (lane & 15);
    _Float16* dst = Bp + (size_t)idx * 16;
#pragma unroll
    for (int e = 0; e < 16; ++e) {
        int k = kt * 32 + hl * 16 + e;
        float v = (n < Ncols) ? W[(size_t)k * Ncols + n] : 0.0f;
        dst[e] = (_Float16)v;
    }
}

// ---------------------------------------------------------------------------
// WMMA GEMM: out[nrows x NCOLS_OUT] = act(A[nrows x KDIM] @ W + bias)
// One wave per 16-row M tile, NTILES 16-col N tiles accumulated in v8f regs.
// The entire packed-B (<= 64 KB) is staged to LDS once per block (4 waves
// share it), then fragments are read via ds_load_b128.
// A fragment (f16 16x32): lanes 0-15 M=0..15, half selects K range:
//   K = kt*32 + 16*half + e  (matches packed B)
// i.e. four contiguous float4 per lane, converted f32->f16 in regs.
// ---------------------------------------------------------------------------
template<int KDIM, int NTILES, int NCOLS_OUT, bool ELU_IN, bool ELU_OUT>
__global__ void wmma_gemm_kernel(const float* __restrict__ A,
                                 const _Float16* __restrict__ Bp,
                                 const float* __restrict__ bias,
                                 float* __restrict__ out, int nrows) {
    constexpr int KT = KDIM / 32;
    constexpr int BHALVES = KT * NTILES * 512;       // halves in packed B
    __shared__ _Float16 Bs[BHALVES];

    // cooperative stage: global -> LDS, 16B chunks (before any wave can exit)
    {
        const uint4* src = (const uint4*)Bp;
        uint4*       dst = (uint4*)Bs;
        const int nchunks = BHALVES / 8;             // 8 halves per 16B
        for (int i = threadIdx.x; i < nchunks; i += blockDim.x) dst[i] = src[i];
    }
    __syncthreads();

    const int lane = threadIdx.x & 31;
    const int wave = threadIdx.x >> 5;
    const int mt   = blockIdx.x * (blockDim.x >> 5) + wave;
    if (mt * 16 >= nrows) return;              // wave-uniform: EXEC stays all-1s
    const int mrow = mt * 16 + (lane & 15);
    const int hl   = lane >> 4;

    v8f acc[NTILES] = {};
#pragma unroll
    for (int kt = 0; kt < KT; ++kt) {
        const float* arow = A + (size_t)mrow * KDIM + kt * 32 + hl * 16;
        v4f f0 = *(const v4f*)(arow);
        v4f f1 = *(const v4f*)(arow + 4);
        v4f f2 = *(const v4f*)(arow + 8);
        v4f f3 = *(const v4f*)(arow + 12);
        float fa[16];
#pragma unroll
        for (int i = 0; i < 4; ++i) {
            fa[i] = f0[i]; fa[4 + i] = f1[i]; fa[8 + i] = f2[i]; fa[12 + i] = f3[i];
        }
        if (ELU_IN) {
#pragma unroll
            for (int i = 0; i < 16; ++i) fa[i] = eluf(fa[i]);
        }
        v16h a;
#pragma unroll
        for (int e = 0; e < 16; ++e) a[e] = (_Float16)fa[e];
#pragma unroll
        for (int nt = 0; nt < NTILES; ++nt) {
            const v16h b = *(const v16h*)(Bs + ((kt * NTILES + nt) * 32 + lane) * 16);
            acc[nt] = __builtin_amdgcn_wmma_f32_16x16x32_f16(
                false, a, false, b, (short)0, acc[nt], false, false);
        }
    }
    // C/D layout: VGPR r -> M = r + 8*half, N = nt*16 + lane%16
#pragma unroll
    for (int nt = 0; nt < NTILES; ++nt) {
#pragma unroll
        for (int r = 0; r < 8; ++r) {
            int row = mt * 16 + r + hl * 8;
            int col = nt * 16 + (lane & 15);
            if (col < NCOLS_OUT) {
                float v = acc[nt][r] + bias[col];
                if (ELU_OUT) v = eluf(v);
                out[(size_t)row * NCOLS_OUT + col] = v;
            }
        }
    }
}

// --------------------------- graph-phase kernels ---------------------------

__global__ void hop0_kernel(const float* __restrict__ h, const float* __restrict__ att0,
                            const float* __restrict__ bias0, float* __restrict__ z,
                            float* __restrict__ zs, float decay0, int n) {
    int idx = blockIdx.x * blockDim.x + threadIdx.x;
    if (idx >= n * HH) return;
    int hd   = idx & (HH - 1);
    int node = idx >> 3;
    size_t off = (size_t)node * (HH * CC) + (size_t)hd * CC;
    const v4f* hv = (const v4f*)(h + off);
    const v4f* av = (const v4f*)(att0 + hd * CC);
    float g = bias0[hd];
#pragma unroll
    for (int q = 0; q < 4; ++q) {
        v4f hq = hv[q], aq = av[q];
#pragma unroll
        for (int j = 0; j < 4; ++j) g += aq[j] * eluf(hq[j]);
    }
#pragma unroll
    for (int q = 0; q < 4; ++q) {
        v4f hq = hv[q], zq, sq;
#pragma unroll
        for (int j = 0; j < 4; ++j) { zq[j] = hq[j] * g; sq[j] = zq[j] * decay0; }
        ((v4f*)(z + off))[q]  = zq;
        ((v4f*)(zs + off))[q] = sq;
    }
}

__global__ void edge_att_kernel(const long long* __restrict__ ei,
                                const float* __restrict__ zs,
                                const float* __restrict__ atts_k,
                                float* __restrict__ a_e, float* __restrict__ deg,
                                int E, int n) {
    int idx = blockIdx.x * blockDim.x + threadIdx.x;
    int Ep = E + n;
    if (idx >= Ep * HH) return;
    int hd = idx & (HH - 1);
    int e  = idx >> 3;
    int r, c;
    if (e < E) { r = (int)ei[e]; c = (int)ei[(size_t)E + e]; }
    else       { r = c = e - E; }                 // self loop
    const v4f* zr = (const v4f*)(zs + (size_t)r * (HH * CC) + (size_t)hd * CC);
    const v4f* zc = (const v4f*)(zs + (size_t)c * (HH * CC) + (size_t)hd * CC);
    const v4f* at = (const v4f*)(atts_k + hd * CC);
    float s = 0.0f;
#pragma unroll
    for (int q = 0; q < 4; ++q) {
        v4f rq = zr[q], cq = zc[q], aq = at[q];
#pragma unroll
        for (int j = 0; j < 4; ++j) s += aq[j] * eluf(rq[j] + cq[j]);
    }
    float a = (s > 20.0f) ? s : log1pf(expf(s));  // softplus
    a += 1e-6f;
    a_e[(size_t)e * HH + hd] = a;
    atomicAdd(&deg[(size_t)c * HH + hd], a);
}

__global__ void dinv_kernel(float* __restrict__ deg, int n) {
    int idx = blockIdx.x * blockDim.x + threadIdx.x;
    if (idx >= n * HH) return;
    float d = deg[idx];
    deg[idx] = (d > 0.0f) ? rsqrtf(d) : 0.0f;
}

__global__ void edge_prop_kernel(const long long* __restrict__ ei,
                                 const float* __restrict__ a_e,
                                 const float* __restrict__ dinv,
                                 const float* __restrict__ h_cur,
                                 float* __restrict__ h_next, int E, int n) {
    int idx = blockIdx.x * blockDim.x + threadIdx.x;
    int Ep = E + n;
    if (idx >= Ep * HH) return;
    int hd = idx & (HH - 1);
    int e  = idx >> 3;
    int r, c;
    if (e < E) { r = (int)ei[e]; c = (int)ei[(size_t)E + e]; }
    else       { r = c = e - E; }
    float w = dinv[(size_t)r * HH + hd] * a_e[(size_t)e * HH + hd]
            * dinv[(size_t)c * HH + hd];
    const v4f* hs = (const v4f*)(h_cur + (size_t)r * (HH * CC) + (size_t)hd * CC);
    float*     hp = h_next + (size_t)c * (HH * CC) + (size_t)hd * CC;
#pragma unroll
    for (int q = 0; q < 4; ++q) {
        v4f hq = hs[q];
#pragma unroll
        for (int j = 0; j < 4; ++j) atomicAdd(&hp[q * 4 + j], w * hq[j]);
    }
}

__global__ void hop_update_kernel(const float* __restrict__ h_next,
                                  const float* __restrict__ hatts_k,
                                  const float* __restrict__ bias_k,
                                  float* __restrict__ z, float* __restrict__ zs,
                                  float decayk, int n) {
    int idx = blockIdx.x * blockDim.x + threadIdx.x;
    if (idx >= n * HH) return;
    int hd   = idx & (HH - 1);
    int node = idx >> 3;
    size_t off = (size_t)node * (HH * CC) + (size_t)hd * CC;
    const v4f* hv = (const v4f*)(h_next + off);
    const v4f* sv = (const v4f*)(zs + off);
    const v4f* a0 = (const v4f*)(hatts_k + hd * 32);
    const v4f* a1 = (const v4f*)(hatts_k + hd * 32 + 16);
    float g = bias_k[hd];
#pragma unroll
    for (int q = 0; q < 4; ++q) {
        v4f hq = hv[q], sq = sv[q], aq = a0[q], bq = a1[q];
#pragma unroll
        for (int j = 0; j < 4; ++j) {
            g += aq[j] * eluf(hq[j]);
            g += bq[j] * eluf(sq[j]);
        }
    }
#pragma unroll
    for (int q = 0; q < 4; ++q) {
        v4f hq = hv[q], zq = ((const v4f*)(z + off))[q], oq, sq;
#pragma unroll
        for (int j = 0; j < 4; ++j) {
            oq[j] = zq[j] + hq[j] * g;
            sq[j] = oq[j] * decayk;
        }
        ((v4f*)(z + off))[q]  = oq;
        ((v4f*)(zs + off))[q] = sq;
    }
}

// ---------------------------------------------------------------------------

extern "C" void kernel_launch(void* const* d_in, const int* in_sizes, int n_in,
                              void* d_out, int out_size, void* d_ws, size_t ws_size,
                              hipStream_t stream) {
    (void)n_in; (void)out_size; (void)ws_size;
    const float*     x    = (const float*)d_in[0];
    const long long* ei   = (const long long*)d_in[1];
    const float*     W0   = (const float*)d_in[2];
    const float*     b0   = (const float*)d_in[3];
    const float*     W1   = (const float*)d_in[4];
    const float*     b1   = (const float*)d_in[5];
    const float*     Wout = (const float*)d_in[6];
    const float*     bout = (const float*)d_in[7];
    const float*     hop_att0   = (const float*)d_in[8];
    const float*     hop_atts   = (const float*)d_in[9];
    const float*     atts       = (const float*)d_in[10];
    const float*     hop_biases = (const float*)d_in[11];
    float* out = (float*)d_out;

    const int N  = in_sizes[0] / 256;
    const int E  = in_sizes[1] / 2;
    const int Ep = E + N;

    // workspace carve-up (256 B aligned)
    char* ws = (char*)d_ws;
    size_t off = 0;
    auto alloc = [&](size_t bytes) -> void* {
        void* p = (void*)(ws + off);
        off += (bytes + 255) & ~(size_t)255;
        return p;
    };
    _Float16* W0p = (_Float16*)alloc((size_t)8 * 8 * 32 * 16 * sizeof(_Float16));
    _Float16* W1p = (_Float16*)alloc((size_t)4 * 8 * 32 * 16 * sizeof(_Float16));
    _Float16* Wop = (_Float16*)alloc((size_t)4 * 3 * 32 * 16 * sizeof(_Float16));
    float* h1  = (float*)alloc((size_t)N * 128 * sizeof(float));
    float* hA  = (float*)alloc((size_t)N * 128 * sizeof(float));
    float* hB  = (float*)alloc((size_t)N * 128 * sizeof(float));
    float* z   = (float*)alloc((size_t)N * 128 * sizeof(float));
    float* zsb = (float*)alloc((size_t)N * 128 * sizeof(float));
    float* a_e = (float*)alloc((size_t)Ep * HH * sizeof(float));
    float* deg = (float*)alloc((size_t)N * HH * sizeof(float));

    float decay[5];
    for (int k = 0; k <= 4; ++k) decay[k] = logf(1.0f / (float)(k + 1) + 1.0f + 1e-6f);

    // pack weights into WMMA-B f16 fragments
    pack_b_kernel<<<(8 * 8 * 32 + 255) / 256, 256, 0, stream>>>(W0,   W0p, 256, 128, 8);
    pack_b_kernel<<<(4 * 8 * 32 + 255) / 256, 256, 0, stream>>>(W1,   W1p, 128, 128, 8);
    pack_b_kernel<<<(4 * 3 * 32 + 255) / 256, 256, 0, stream>>>(Wout, Wop, 128,  40, 3);

    const int mtiles  = (N + 15) / 16;
    const int gblocks = (mtiles + 3) / 4;   // 4 waves (M tiles) per 128-thread block

    // h1 = elu(x @ W0 + b0) ; hA = h1 @ W1 + b1
    wmma_gemm_kernel<256, 8, 128, false, true ><<<gblocks, 128, 0, stream>>>(x,  W0p, b0, h1, N);
    wmma_gemm_kernel<128, 8, 128, false, false><<<gblocks, 128, 0, stream>>>(h1, W1p, b1, hA, N);

    const int nh = N * HH;
    hop0_kernel<<<(nh + 255) / 256, 256, 0, stream>>>(hA, hop_att0, hop_biases, z, zsb,
                                                      decay[0], N);

    float* hcur = hA;
    float* hnxt = hB;
    const int eh = Ep * HH;
    for (int k = 1; k <= 4; ++k) {
        hipMemsetAsync(deg,  0, (size_t)N * HH  * sizeof(float), stream);
        hipMemsetAsync(hnxt, 0, (size_t)N * 128 * sizeof(float), stream);
        edge_att_kernel<<<(eh + 255) / 256, 256, 0, stream>>>(
            ei, zsb, atts + (size_t)(k - 1) * HH * CC, a_e, deg, E, N);
        dinv_kernel<<<(nh + 255) / 256, 256, 0, stream>>>(deg, N);
        edge_prop_kernel<<<(eh + 255) / 256, 256, 0, stream>>>(
            ei, a_e, deg, hcur, hnxt, E, N);
        hop_update_kernel<<<(nh + 255) / 256, 256, 0, stream>>>(
            hnxt, hop_atts + (size_t)(k - 1) * HH * 32, hop_biases + (size_t)k * HH,
            z, zsb, decay[k], N);
        float* t = hcur; hcur = hnxt; hnxt = t;
    }

    // out = elu(z) @ Wout + bout   (40 cols padded to 3 N-tiles, bound-checked store)
    wmma_gemm_kernel<128, 3, 40, true, false><<<gblocks, 128, 0, stream>>>(z, Wop, bout, out, N);
}
